// SchNetGNN_25623774888014
// MI455X (gfx1250) — compile-verified, matrix operand order checked by reading, not run on hardware
//
#include <hip/hip_runtime.h>

typedef __attribute__((ext_vector_type(16))) _Float16 v16h;
typedef __attribute__((ext_vector_type(8)))  _Float16 h8;
typedef __attribute__((ext_vector_type(4)))  _Float16 h4;
typedef __attribute__((ext_vector_type(8)))  float    v8f;

#define V_NODES 100000
#define E_EDGES 1600000
#define DF      64
#define NC      300
#define NCP     320      // centers padded to 10 K-chunks of 32
#define LAYERS  3
#define EPB     128      // edges per block (E_EDGES % EPB == 0)
#define LDH     72       // LDS half-stride: 144B = 9*16B -> 16B-aligned frag chunks

#define WMMA16(A,B,C) __builtin_amdgcn_wmma_f32_16x16x32_f16(false,(A),false,(B),(short)0,(C),false,false)

#define LN2   0.69314718055994530942f
#define L2E   1.44269504088896340736f
#define RSTEP 0.100334448160535117f      // 30/299 (np.linspace spacing)
#define KEXP  (-10.0f * L2E)             // gamma=10, folded into exp2

// ---------------------------------------------------------------------------
// helpers
// ---------------------------------------------------------------------------
__device__ __forceinline__ float sspf(float x) {
  // softplus(x) - ln2 with raw HW transcendentals (arg of log is in [1,2]:
  // no denorm/extended-precision fixups needed).
  const float t  = __builtin_amdgcn_exp2f(-fabsf(x) * L2E);   // e^-|x|
  const float lg = __builtin_amdgcn_logf(1.0f + t);           // log2(1+t)
  return fmaxf(x, 0.0f) + __builtin_fmaf(LN2, lg, -LN2);
}

// A fragment (16x32 f16, MxK) from row-major LDS with stride LDH (16B aligned).
// ISA 7.12.2: lanes 0-15: elems 0..7 -> K 0..7, 8..15 -> K 16..23 (m = lane)
//             lanes16-31: elems 0..7 -> K 8..15, 8..15 -> K 24..31
__device__ __forceinline__ v16h load_a_frag72(const _Float16* __restrict__ base,
                                              int k_base, int lane) {
  const int m  = lane & 15;
  const int kh = (lane >> 4) * 8;
  const _Float16* row = base + m * LDH + k_base + kh;
  h8 lo = *(const h8*)(row);        // ds_load_b128
  h8 hi = *(const h8*)(row + 16);   // ds_load_b128
  v16h a;
#pragma unroll
  for (int i = 0; i < 8; ++i) { a[i] = lo[i]; a[8 + i] = hi[i]; }
  return a;
}

// B fragment from pre-swizzled table: frag-major, each lane's 16 halves
// contiguous (32B). Works for LDS or global pointers.
__device__ __forceinline__ v16h load_b_swz(const _Float16* __restrict__ Ws,
                                           int frag, int lane) {
  const h8* p = (const h8*)(Ws + ((size_t)frag * 32 + lane) * 16);
  h8 lo = p[0];
  h8 hi = p[1];
  v16h b;
#pragma unroll
  for (int i = 0; i < 8; ++i) { b[i] = lo[i]; b[8 + i] = hi[i]; }
  return b;
}

// RBF A-fragment generated in registers. No k<300 guard needed: W1 rows
// k>=300 are zeroed, and exp2(KEXP*r*r) is always finite for d in [0,30].
__device__ __forceinline__ v16h rbf_a_frag(float d, int k_base, int lane) {
  const int kh = (lane >> 4) * 8;
  const float rb = d - (float)(k_base + kh) * RSTEP;  // r at elem group start
  v16h a;
#pragma unroll
  for (int i = 0; i < 8; ++i) {
    const float r0 = rb - RSTEP * (float)i;            // K = base+kh+i
    const float r1 = rb - RSTEP * (float)(16 + i);     // K = base+16+kh+i
    a[i]     = (_Float16)__builtin_amdgcn_exp2f(KEXP * r0 * r0);
    a[8 + i] = (_Float16)__builtin_amdgcn_exp2f(KEXP * r1 * r1);
  }
  return a;
}

// ---------------------------------------------------------------------------
// weight prep: f32 -> f16, swizzled into B-fragment order.
// For frag f = kc*4+ct, lane l, elem e:
//   kh = (l>>4)*8 ; k = kc*32 + (e<8 ? kh+e : 16+kh+e-8) ; n = ct*16 + (l&15)
// ---------------------------------------------------------------------------
__global__ __launch_bounds__(256) void prep_weights_kernel(
    const float* __restrict__ eW1, const float* __restrict__ eW2,
    const float* __restrict__ nW,  const float* __restrict__ cW,
    const float* __restrict__ oW,
    _Float16* __restrict__ W1s, _Float16* __restrict__ W2s,
    _Float16* __restrict__ nWs, _Float16* __restrict__ cWs,
    _Float16* __restrict__ oWs)
{
  const int idx = blockIdx.x * 256 + threadIdx.x;

  if (idx < LAYERS * NCP * DF) {          // W1: K padded 300 -> 320 (zeros)
    const int l = idx / (NCP * DF);
    const int r = idx % (NCP * DF);
    const int frag = r >> 9, lane = (r >> 4) & 31, e = r & 15;
    const int kc = frag >> 2, ct = frag & 3;
    const int kh = (lane >> 4) * 8;
    const int k  = kc * 32 + ((e < 8) ? (kh + e) : (16 + kh + e - 8));
    const int n  = ct * 16 + (lane & 15);
    W1s[idx] = (_Float16)(k < NC ? eW1[((size_t)l * NC + k) * DF + n] : 0.0f);
  }
  if (idx < LAYERS * DF * DF) {           // 64x64 weights
    const int l = idx / (DF * DF);
    const int r = idx % (DF * DF);
    const int frag = r >> 9, lane = (r >> 4) & 31, e = r & 15;
    const int kc = frag >> 2, ct = frag & 3;
    const int kh = (lane >> 4) * 8;
    const int k  = kc * 32 + ((e < 8) ? (kh + e) : (16 + kh + e - 8));
    const int n  = ct * 16 + (lane & 15);
    const size_t s = ((size_t)l * DF + k) * DF + n;
    W2s[idx] = (_Float16)eW2[s];
    nWs[idx] = (_Float16)nW[s];
    cWs[idx] = (_Float16)cW[s];
    oWs[idx] = (_Float16)oW[s];
  }
}

// ---------------------------------------------------------------------------
// embedding lookup (float4 vectorized): h[v,:] = embed[type[v],:]
// ---------------------------------------------------------------------------
__global__ __launch_bounds__(256) void embed_kernel(
    const int* __restrict__ types, const float* __restrict__ embed,
    float* __restrict__ h)
{
  const int idx = blockIdx.x * 256 + threadIdx.x;       // V * 16 total
  const int v = idx >> 4, c4 = idx & 15;
  ((float4*)h)[idx] = ((const float4*)embed)[types[v] * 16 + c4];
}

// ---------------------------------------------------------------------------
// node-side GEMM: Y = (ssp?)(X[M,64] @ W[64,64] + b)
// 256 thr = 8 waves, tile 32 rows x 64 cols. B frags read straight from the
// L2-resident swizzled global table (coalesced 32B/lane) -> no B LDS staging.
// ---------------------------------------------------------------------------
__global__ __launch_bounds__(256) void node_gemm_kernel(
    const float* __restrict__ X, const _Float16* __restrict__ Wswz,
    const float* __restrict__ bias, float* __restrict__ Y, int apply_ssp)
{
  __shared__ __align__(16) _Float16 sX[32 * LDH];

  const int tid  = threadIdx.x;
  const int row0 = blockIdx.x * 32;

  // stage X tile as f16, float4 global reads -> 8B LDS vector stores
  for (int i = tid; i < 32 * 16; i += 256) {
    const int r = i >> 4, q = i & 15;
    const float4 v = ((const float4*)X)[(size_t)(row0 + r) * 16 + q];
    h4 hv4 = { (_Float16)v.x, (_Float16)v.y, (_Float16)v.z, (_Float16)v.w };
    *(h4*)(sX + r * LDH + q * 4) = hv4;
  }
  __syncthreads();

  const int lane = tid & 31, wave = tid >> 5;
  const int rt = wave & 1;          // row tile 0..1
  const int ct = wave >> 1;         // col tile 0..3

  v8f acc = {};
#pragma unroll
  for (int kc = 0; kc < 2; ++kc) {
    v16h a = load_a_frag72(sX + rt * 16 * LDH, kc * 32, lane);
    v16h b = load_b_swz(Wswz, kc * 4 + ct, lane);
    acc = WMMA16(a, b, acc);
  }

  const int n  = ct * 16 + (lane & 15);
  const int m0 = rt * 16 + (lane >> 4) * 8;
  const float bv = bias[n];
#pragma unroll
  for (int j = 0; j < 8; ++j) {
    float v = acc[j] + bv;
    if (apply_ssp) v = sspf(v);
    Y[(size_t)(row0 + m0 + j) * DF + n] = v;
  }
}

// ---------------------------------------------------------------------------
// fused CFConv edge kernel, 128 edges per block, 8 waves:
// wave w owns edge rows [16w,16w+16), all 4 feature col-tiles.
//   rbf (in-register A frags) -> 40x WMMA MLP1 -> ssp
//   -> 8x WMMA MLP2 -> ssp -> msg = hv[src]*he -> atomic scatter to agg[dst]
// ---------------------------------------------------------------------------
__global__ __launch_bounds__(256) void edge_cfconv_kernel(
    const float* __restrict__ dists, const int* __restrict__ src,
    const int* __restrict__ dst,
    const _Float16* __restrict__ W1swz,   // [NCP/32*4 frags][32][16] f16
    const _Float16* __restrict__ W2swz,   // [8 frags][32][16] f16
    const float* __restrict__ b1, const float* __restrict__ b2,
    const float* __restrict__ hv, float* __restrict__ agg)
{
  __shared__ __align__(16) _Float16 sW1[NCP * DF];      // 40 KB (swizzled)
  __shared__ __align__(16) _Float16 sW2[DF * DF];       // 8 KB  (swizzled)
  __shared__ __align__(16) _Float16 sH[EPB * LDH];      // 18 KB (he1, then he)
  __shared__ float sD[EPB];
  __shared__ int   sS[EPB];
  __shared__ int   sT[EPB];

  const int tid = threadIdx.x;
  const int e0  = blockIdx.x * EPB;

  if (tid < EPB) {
    sD[tid] = dists[e0 + tid];
    sS[tid] = src[e0 + tid];
    sT[tid] = dst[e0 + tid];
  }
  // stage swizzled weights with 16B vector copies
  {
    const uint4* g1 = (const uint4*)W1swz;
    uint4* l1 = (uint4*)sW1;
    for (int i = tid; i < NCP * DF / 8; i += 256) l1[i] = g1[i];
    const uint4* g2 = (const uint4*)W2swz;
    uint4* l2 = (uint4*)sW2;
    for (int i = tid; i < DF * DF / 8; i += 256) l2[i] = g2[i];
  }
  __syncthreads();

  const int lane = tid & 31;
  const int rt   = tid >> 5;                 // wave = row tile 0..7
  const int nn   = lane & 15;
  const int m0   = rt * 16 + (lane >> 4) * 8;
  const float dme = sD[rt * 16 + nn];

  // ---- MLP1: [128 x 320] @ [320 x 64], A generated on the fly ----
  v8f acc[4] = {};
#pragma unroll
  for (int kc = 0; kc < NCP / 32; ++kc) {
    v16h a = rbf_a_frag(dme, kc * 32, lane);
#pragma unroll
    for (int ct = 0; ct < 4; ++ct) {
      v16h b = load_b_swz(sW1, kc * 4 + ct, lane);
      acc[ct] = WMMA16(a, b, acc[ct]);
    }
  }
#pragma unroll
  for (int ct = 0; ct < 4; ++ct) {
    const int c = ct * 16 + nn;
    const float bb = b1[c];
#pragma unroll
    for (int j = 0; j < 8; ++j)
      sH[(m0 + j) * LDH + c] = (_Float16)sspf(acc[ct][j] + bb);
  }
  __syncthreads();

  // ---- MLP2: [128 x 64] @ [64 x 64] ----
  v8f a2[4] = {};
#pragma unroll
  for (int kc = 0; kc < 2; ++kc) {
    v16h a = load_a_frag72(sH + rt * 16 * LDH, kc * 32, lane);
#pragma unroll
    for (int ct = 0; ct < 4; ++ct) {
      v16h b = load_b_swz(sW2, kc * 4 + ct, lane);
      a2[ct] = WMMA16(a, b, a2[ct]);
    }
  }
  float r[4][8];
#pragma unroll
  for (int ct = 0; ct < 4; ++ct) {
    const float bb = b2[ct * 16 + nn];
#pragma unroll
    for (int j = 0; j < 8; ++j) r[ct][j] = sspf(a2[ct][j] + bb);
  }
  __syncthreads();                 // all reads of sH (he1) complete
#pragma unroll
  for (int ct = 0; ct < 4; ++ct) {
    const int c = ct * 16 + nn;
#pragma unroll
    for (int j = 0; j < 8; ++j) sH[(m0 + j) * LDH + c] = (_Float16)r[ct][j];
  }
  __syncthreads();                 // sH now holds he

  // ---- gather hv[src], multiply, atomic scatter-add into agg[dst] ----
  const int f = tid & 63;
  for (int e = tid >> 6; e < EPB; e += 4) {
    const float m = hv[(size_t)sS[e] * DF + f] * (float)sH[e * LDH + f];
    atomicAdd(&agg[(size_t)sT[e] * DF + f], m);
  }
}

// ---------------------------------------------------------------------------
// driver
// ---------------------------------------------------------------------------
extern "C" void kernel_launch(void* const* d_in, const int* in_sizes, int n_in,
                              void* d_out, int out_size, void* d_ws, size_t ws_size,
                              hipStream_t stream)
{
  const int*   node_types = (const int*)  d_in[0];
  const float* edge_dists = (const float*)d_in[1];
  const int*   src        = (const int*)  d_in[2];
  const int*   dst        = (const int*)  d_in[3];
  const float* embed      = (const float*)d_in[4];
  const float* edge_W1    = (const float*)d_in[5];
  const float* edge_b1    = (const float*)d_in[6];
  const float* edge_W2    = (const float*)d_in[7];
  const float* edge_b2    = (const float*)d_in[8];
  const float* node_W     = (const float*)d_in[9];
  const float* node_b     = (const float*)d_in[10];
  const float* cfout_W    = (const float*)d_in[11];
  const float* cfout_b    = (const float*)d_in[12];
  const float* out_W      = (const float*)d_in[13];
  const float* out_b      = (const float*)d_in[14];
  float* out = (float*)d_out;

  // workspace carve-up (256B aligned)
  char* ws = (char*)d_ws;
  size_t off = 0;
  auto take = [&](size_t bytes) -> void* {
    void* p = ws + off;
    off = (off + bytes + 255) & ~(size_t)255;
    return p;
  };
  _Float16* W1s = (_Float16*)take((size_t)LAYERS * NCP * DF * 2);
  _Float16* W2s = (_Float16*)take((size_t)LAYERS * DF * DF * 2);
  _Float16* nWs = (_Float16*)take((size_t)LAYERS * DF * DF * 2);
  _Float16* cWs = (_Float16*)take((size_t)LAYERS * DF * DF * 2);
  _Float16* oWs = (_Float16*)take((size_t)LAYERS * DF * DF * 2);
  float* hA  = (float*)take((size_t)V_NODES * DF * 4);
  float* hv  = (float*)take((size_t)V_NODES * DF * 4);
  float* agg = (float*)take((size_t)V_NODES * DF * 4);
  if (off > ws_size) return;  // insufficient workspace (deterministic no-op)

  // one-time prep
  prep_weights_kernel<<<(LAYERS * NCP * DF + 255) / 256, 256, 0, stream>>>(
      edge_W1, edge_W2, node_W, cfout_W, out_W, W1s, W2s, nWs, cWs, oWs);
  embed_kernel<<<(V_NODES * 16) / 256, 256, 0, stream>>>(node_types, embed, hA);

  for (int l = 0; l < LAYERS; ++l) {
    hipMemsetAsync(agg, 0, (size_t)V_NODES * DF * 4, stream);

    // hv = h @ node_W + node_b
    node_gemm_kernel<<<V_NODES / 32, 256, 0, stream>>>(
        hA, nWs + (size_t)l * DF * DF, node_b + l * DF, hv, 0);

    // fused rbf -> edge MLP -> gather/mul/scatter
    edge_cfconv_kernel<<<E_EDGES / EPB, 256, 0, stream>>>(
        edge_dists, src, dst,
        W1s + (size_t)l * NCP * DF, W2s + (size_t)l * DF * DF,
        edge_b1 + l * DF, edge_b2 + l * DF, hv, agg);

    // agg = ssp(agg @ cfout_W + b)   (safe in-place: block reads its rows first)
    node_gemm_kernel<<<V_NODES / 32, 256, 0, stream>>>(
        agg, cWs + (size_t)l * DF * DF, cfout_b + l * DF, agg, 1);

    // h = agg @ out_W + b  (last layer writes final output)
    float* yout = (l == LAYERS - 1) ? out : hA;
    node_gemm_kernel<<<V_NODES / 32, 256, 0, stream>>>(
        agg, oWs + (size_t)l * DF * DF, out_b + l * DF, yout, 0);
  }
}
